// EquivariantDynamics_54082228191600
// MI455X (gfx1250) — matrix operand order repaired and checked by reading, not compiled, not run on hardware
//
#include <hip/hip_runtime.h>
#include <hip/hip_bf16.h>
#include <math.h>

// ---------------------------------------------------------------------------
// EGNN (EquivariantDynamics) for MI455X / gfx1250, wave32 + WMMA f16->f32.
//
// Pipeline (all on `stream`):
//   1. frag_convert  : f32 weights -> f16 WMMA-B fragment-major layout in ws
//   2. node_init     : h0 = concat(feats) @ Wh + bh   (VALU, tiny GEMM)
//   3. edge_dist     : a[e] = ||x[dst]-x[src]||^2 (from initial coords)
//   4. per layer l:
//        memcpy x_cur -> x_next
//        edge_msg     : per 16-edge tile per wave:
//                         m1 = silu(h_s@W1a + h_d@W1b + a*w1c + be1)   (WMMA)
//                         m  = silu(m1@We2 + be2)                      (WMMA)
//                         u  = silu(m@Wx1 + bx1); w = u@Wx2            (WMMA+red)
//                         atomicAdd x_next[dst] += (dx/(|dx|+1))*w
//                         atomicAdd agg[dst]    += m      (layer 0)
//        node_update  : h += silu([h,agg]@Wh1+bh1)@Wh2 + bh2           (WMMA)
//   5. memcpy x_final -> d_out
// ---------------------------------------------------------------------------

#define HF 128
#define NFEAT 139

typedef _Float16 h16;
typedef __attribute__((ext_vector_type(16))) _Float16 v16h;
typedef __attribute__((ext_vector_type(8)))  float    v8f;

union HV16 { v16h v; uint4 q[2]; h16 h[16]; };

__device__ __forceinline__ float silu_f(float x) { return x / (1.0f + __expf(-x)); }

__device__ __forceinline__ v8f wmma16(v16h a, v16h b, v8f c) {
  // D(f32 16x16) = A(f16 16x32) * B(f16 32x16) + C
  return __builtin_amdgcn_wmma_f32_16x16x32_f16(false, a, false, b, (short)0, c,
                                                false, false);
}

// B fragment, fragment-major in LDS: (((c*8 + t)*32 + lane)*16) halfs, 32B aligned.
__device__ __forceinline__ const v16h* bfrag_ptr(const h16* base, int c, int t, int lane) {
  return (const v16h*)(base + ((((c << 3) + t) << 5) + lane) * 16);
}

// A fragment from a global f16 row (row-major 128 halfs).
// ISA 16-bit A layout: lanes 0-15 hold K = c*32 + {0..7, 16..23};
//                      lanes 16-31 hold K = c*32 + {8..15, 24..31}.
__device__ __forceinline__ v16h afrag_gh(const h16* row, int c, int lane) {
  HV16 u;
  const int base = (c << 5) + ((lane & 16) ? 8 : 0);
  u.q[0] = *(const uint4*)(row + base);
  u.q[1] = *(const uint4*)(row + base + 16);
  return u.v;
}

// A fragment from a global f32 row (convert to f16 on the fly).
__device__ __forceinline__ v16h afrag_gf(const float* row, int c, int lane) {
  HV16 u;
  const int base = (c << 5) + ((lane & 16) ? 8 : 0);
  const float4 f0 = *(const float4*)(row + base);
  const float4 f1 = *(const float4*)(row + base + 4);
  const float4 f2 = *(const float4*)(row + base + 16);
  const float4 f3 = *(const float4*)(row + base + 20);
  u.h[0]=(h16)f0.x; u.h[1]=(h16)f0.y; u.h[2]=(h16)f0.z; u.h[3]=(h16)f0.w;
  u.h[4]=(h16)f1.x; u.h[5]=(h16)f1.y; u.h[6]=(h16)f1.z; u.h[7]=(h16)f1.w;
  u.h[8]=(h16)f2.x; u.h[9]=(h16)f2.y; u.h[10]=(h16)f2.z; u.h[11]=(h16)f2.w;
  u.h[12]=(h16)f3.x; u.h[13]=(h16)f3.y; u.h[14]=(h16)f3.z; u.h[15]=(h16)f3.w;
  return u.v;
}

// A fragment from a per-wave LDS tile [16][128] halfs.
__device__ __forceinline__ v16h afrag_lds(const h16* tile, int c, int lane) {
  HV16 u;
  const int base = (lane & 15) * HF + (c << 5) + ((lane & 16) ? 8 : 0);
  u.q[0] = *(const uint4*)(tile + base);
  u.q[1] = *(const uint4*)(tile + base + 16);
  return u.v;
}

// ---------------------------------------------------------------------------
// Weight fragment conversion: W[k][n] f32 -> fragment-major f16.
// idx = ((c*8 + t)*32 + lane)*16 + j ; element = W[c*32 + (lane&16?8:0) + (j&7) + (j&8?16:0)][t*16 + (lane&15)]
// ---------------------------------------------------------------------------
__global__ void frag_convert_kernel(const float* __restrict__ W, h16* __restrict__ frag, int K) {
  const int idx = blockIdx.x * blockDim.x + threadIdx.x;
  if (idx >= K * HF) return;
  const int j    = idx & 15;
  const int lane = (idx >> 4) & 31;
  const int t    = (idx >> 9) & 7;
  const int c    = idx >> 12;
  const int k    = (c << 5) + ((lane & 16) ? 8 : 0) + (j & 7) + ((j & 8) << 1);
  const int col  = (t << 4) + (lane & 15);
  frag[idx] = (h16)W[k * HF + col];
}

// ---------------------------------------------------------------------------
// h0 = f @ Wh + bh (NF=139 -> H=128). One node per block, one col per thread.
// ---------------------------------------------------------------------------
__global__ void node_init_kernel(
    const int* __restrict__ atom, const float* __restrict__ temb,
    const float* __restrict__ masses, const float* __restrict__ mnorm,
    const float* __restrict__ clab, const float* __restrict__ cmask,
    const float* __restrict__ mom, const float* __restrict__ embed,
    const float* __restrict__ Wh, const float* __restrict__ bh,
    float* __restrict__ h, h16* __restrict__ hhalf, int nN) {
  __shared__ float f[NFEAT];
  const int n = blockIdx.x;
  if (n >= nN) return;
  const int t = threadIdx.x;
  for (int k = t; k < NFEAT; k += blockDim.x) {
    float v;
    if (k < 64)        v = embed[atom[n] * 64 + k];
    else if (k < 128)  v = temb[(size_t)n * 64 + (k - 64)];
    else if (k == 128) v = masses[n] * (1.0f / 12.0f);
    else if (k == 129) v = mnorm[n];
    else if (k < 133)  v = clab[n * 3 + (k - 130)];
    else if (k < 136)  v = cmask[n * 3 + (k - 133)];
    else               v = mom[n * 3 + (k - 136)];
    f[k] = v;
  }
  __syncthreads();
  float acc = bh[t];
  for (int k = 0; k < NFEAT; ++k) acc = fmaf(f[k], Wh[k * HF + t], acc);
  h[(size_t)n * HF + t]     = acc;
  hhalf[(size_t)n * HF + t] = (h16)acc;
}

// ---------------------------------------------------------------------------
// a[e] = ||x[dst]-x[src]||^2 from initial coordinates.
// ---------------------------------------------------------------------------
__global__ void edge_dist_kernel(const float* __restrict__ x, const int* __restrict__ src,
                                 const int* __restrict__ dst, float* __restrict__ a, int nE) {
  const int e = blockIdx.x * blockDim.x + threadIdx.x;
  if (e >= nE) return;
  const int s = src[e], d = dst[e];
  const float d0 = x[d * 3 + 0] - x[s * 3 + 0];
  const float d1 = x[d * 3 + 1] - x[s * 3 + 1];
  const float d2 = x[d * 3 + 2] - x[s * 3 + 2];
  a[e] = d0 * d0 + d1 * d1 + d2 * d2;
}

// ---------------------------------------------------------------------------
// Edge message + coordinate update kernel. 128 threads = 4 waves; one wave
// owns a 16-edge tile; weights (4 x 128x128 f16 frags = 128 KiB) staged to LDS.
// ---------------------------------------------------------------------------
#define EDGE_WTS_BYTES (4 * 16384 * 2)        // 131072
#define PW_BYTES 9216                         // per-wave scratch region
#define EDGE_SMEM (EDGE_WTS_BYTES + 4 * PW_BYTES)

__launch_bounds__(128, 1)
__global__ void edge_msg_kernel(
    const h16* __restrict__ hhalf, const float* __restrict__ xin,
    float* __restrict__ xout, float* __restrict__ agg,
    const int* __restrict__ src, const int* __restrict__ dst,
    const float* __restrict__ aBuf, const h16* __restrict__ wfrag,
    const float* __restrict__ be1, const float* __restrict__ be2,
    const float* __restrict__ bx1, const float* __restrict__ w1c,
    const float* __restrict__ wx2, int nE, int doAgg) {
  extern __shared__ unsigned char smem[];
  h16* wts = (h16*)smem;  // [4 mats][4c][8t][32 lane][16] halfs
  const int lane = threadIdx.x & 31;
  const int wave = threadIdx.x >> 5;
  unsigned char* pw = smem + EDGE_WTS_BYTES + wave * PW_BYTES;
  h16*   tile = (h16*)pw;             // [16][128] f16 activation tile (4096 B)
  float* aT   = (float*)(pw + 4096);  // a[e] per row
  int*   dT   = (int*)(pw + 4160);    // dst per row
  float* wred = (float*)(pw + 4224);  // [16][16] w-dot partials

  {  // stage fragment-major weights to LDS
    const uint4* g = (const uint4*)wfrag;
    uint4* s = (uint4*)wts;
    for (int i = threadIdx.x; i < EDGE_WTS_BYTES / 16; i += 128) s[i] = g[i];
  }
  __syncthreads();

  const int m   = lane & 15;
  const int hi8 = (lane & 16) ? 8 : 0;

  // lane-constant per-column values (hoisted out of the tile loop)
  float be1v[8], be2v[8], bx1v[8], w1cv[8], wx2v[8];
#pragma unroll
  for (int t = 0; t < 8; ++t) {
    be1v[t] = be1[t * 16 + m];
    be2v[t] = be2[t * 16 + m];
    bx1v[t] = bx1[t * 16 + m];
    w1cv[t] = w1c[t * 16 + m];
    wx2v[t] = wx2[t * 16 + m];
  }

  const int nT   = (nE + 15) >> 4;
  const int wgid = blockIdx.x * 4 + wave;
  const int wstr = gridDim.x * 4;

  for (int ti = wgid; ti < nT; ti += wstr) {
    const int  e  = ti * 16 + m;
    const bool ev = (e < nE);
    const int  ec = ev ? e : (nE - 1);
    const int  s  = src[ec];
    const int  d  = dst[ec];
    const h16* hs = hhalf + (size_t)s * HF;
    const h16* hd = hhalf + (size_t)d * HF;
    if (lane < 16) { aT[m] = ev ? aBuf[ec] : 0.0f; dT[m] = d; }
    asm volatile("s_wait_dscnt 0x0" ::: "memory");

    // ---- m1 = silu(h_s@W1a + h_d@W1b + a*w1c + be1) ----
    v8f acc[8];
#pragma unroll
    for (int t = 0; t < 8; ++t)
#pragma unroll
      for (int r = 0; r < 8; ++r) acc[t][r] = be1v[t];
#pragma unroll
    for (int c = 0; c < 4; ++c) {
      const v16h as = afrag_gh(hs, c, lane);
      const v16h ad = afrag_gh(hd, c, lane);
#pragma unroll
      for (int t = 0; t < 8; ++t) {
        acc[t] = wmma16(as, *bfrag_ptr(wts,           c, t, lane), acc[t]);
        acc[t] = wmma16(ad, *bfrag_ptr(wts + 16384,   c, t, lane), acc[t]);
      }
    }
#pragma unroll
    for (int t = 0; t < 8; ++t)
#pragma unroll
      for (int r = 0; r < 8; ++r) {
        const float v = silu_f(acc[t][r] + aT[r + hi8] * w1cv[t]);
        tile[(r + hi8) * HF + t * 16 + m] = (h16)v;
      }
    asm volatile("s_wait_dscnt 0x0" ::: "memory");

    // ---- m = silu(m1 @ We2 + be2) ----
#pragma unroll
    for (int t = 0; t < 8; ++t)
#pragma unroll
      for (int r = 0; r < 8; ++r) acc[t][r] = be2v[t];
#pragma unroll
    for (int c = 0; c < 4; ++c) {
      const v16h am = afrag_lds(tile, c, lane);
#pragma unroll
      for (int t = 0; t < 8; ++t)
        acc[t] = wmma16(am, *bfrag_ptr(wts + 2 * 16384, c, t, lane), acc[t]);
    }
#pragma unroll
    for (int t = 0; t < 8; ++t)
#pragma unroll
      for (int r = 0; r < 8; ++r)
        tile[(r + hi8) * HF + t * 16 + m] = (h16)silu_f(acc[t][r]);
    asm volatile("s_wait_dscnt 0x0" ::: "memory");

    // ---- u = silu(m @ Wx1 + bx1);  w = u @ Wx2 ----
#pragma unroll
    for (int t = 0; t < 8; ++t)
#pragma unroll
      for (int r = 0; r < 8; ++r) acc[t][r] = bx1v[t];
#pragma unroll
    for (int c = 0; c < 4; ++c) {
      const v16h am = afrag_lds(tile, c, lane);
#pragma unroll
      for (int t = 0; t < 8; ++t)
        acc[t] = wmma16(am, *bfrag_ptr(wts + 3 * 16384, c, t, lane), acc[t]);
    }
    float p[8];
#pragma unroll
    for (int r = 0; r < 8; ++r) p[r] = 0.0f;
#pragma unroll
    for (int t = 0; t < 8; ++t)
#pragma unroll
      for (int r = 0; r < 8; ++r) p[r] += silu_f(acc[t][r]) * wx2v[t];
#pragma unroll
    for (int r = 0; r < 8; ++r) wred[(r + hi8) * 16 + m] = p[r];
    asm volatile("s_wait_dscnt 0x0" ::: "memory");

    // ---- coordinate scatter: x_next[dst] += (dx/(|dx|+1)) * w ----
    if (lane < 16 && ev) {
      float w = 0.0f;
#pragma unroll
      for (int i = 0; i < 16; ++i) w += wred[m * 16 + i];
      const float q0 = xin[(size_t)d * 3 + 0] - xin[(size_t)s * 3 + 0];
      const float q1 = xin[(size_t)d * 3 + 1] - xin[(size_t)s * 3 + 1];
      const float q2 = xin[(size_t)d * 3 + 2] - xin[(size_t)s * 3 + 2];
      const float kk = w / (sqrtf(q0 * q0 + q1 * q1 + q2 * q2) + 1.0f);
      atomicAdd(&xout[(size_t)d * 3 + 0], q0 * kk);
      atomicAdd(&xout[(size_t)d * 3 + 1], q1 * kk);
      atomicAdd(&xout[(size_t)d * 3 + 2], q2 * kk);
    }

    // ---- agg scatter (layer 0 only): agg[dst] += m ----
    if (doAgg) {
      for (int idx = lane; idx < 16 * HF; idx += 32) {
        const int row = idx >> 7;
        if (ti * 16 + row < nE)
          atomicAdd(&agg[(size_t)dT[row] * HF + (idx & 127)], (float)tile[idx]);
      }
    }
  }
}

// ---------------------------------------------------------------------------
// Hidden-state update: h += silu([h,agg]@Wh1 + bh1)@Wh2 + bh2 ; refresh hhalf.
// ---------------------------------------------------------------------------
#define NODE_WTS_BYTES ((32768 + 16384) * 2)  // 98304
#define NODE_SMEM (NODE_WTS_BYTES + 4 * PW_BYTES)

__launch_bounds__(128, 1)
__global__ void node_update_kernel(
    float* __restrict__ hbuf, h16* __restrict__ hhalf,
    const float* __restrict__ agg,
    const h16* __restrict__ wf1, const h16* __restrict__ wf2,
    const float* __restrict__ bh1, const float* __restrict__ bh2, int nN) {
  extern __shared__ unsigned char smem[];
  h16* w1 = (h16*)smem;             // Wh1 frags, K=256 (65536 B)
  h16* w2 = (h16*)(smem + 65536);   // Wh2 frags, K=128 (32768 B)
  const int lane = threadIdx.x & 31;
  const int wave = threadIdx.x >> 5;
  unsigned char* pw = smem + NODE_WTS_BYTES + wave * PW_BYTES;
  h16*   tile  = (h16*)pw;    // f16 tile (phase 1 output)
  float* tileF = (float*)pw;  // f32 tile (phase 2 output; reuses region)

  {
    const uint4* g1 = (const uint4*)wf1;
    uint4* s1 = (uint4*)w1;
    for (int i = threadIdx.x; i < 65536 / 16; i += 128) s1[i] = g1[i];
    const uint4* g2 = (const uint4*)wf2;
    uint4* s2 = (uint4*)w2;
    for (int i = threadIdx.x; i < 32768 / 16; i += 128) s2[i] = g2[i];
  }
  __syncthreads();

  const int m   = lane & 15;
  const int hi8 = (lane & 16) ? 8 : 0;
  float bh1v[8];
#pragma unroll
  for (int t = 0; t < 8; ++t) bh1v[t] = bh1[t * 16 + m];

  const int nT   = (nN + 15) >> 4;
  const int wgid = blockIdx.x * 4 + wave;
  const int wstr = gridDim.x * 4;

  for (int ti = wgid; ti < nT; ti += wstr) {
    const int n  = ti * 16 + m;
    const int nc = (n < nN) ? n : (nN - 1);
    const h16*   hr = hhalf + (size_t)nc * HF;
    const float* ar = agg + (size_t)nc * HF;

    v8f acc[8];
#pragma unroll
    for (int t = 0; t < 8; ++t)
#pragma unroll
      for (int r = 0; r < 8; ++r) acc[t][r] = bh1v[t];
#pragma unroll
    for (int c = 0; c < 8; ++c) {
      const v16h af = (c < 4) ? afrag_gh(hr, c, lane) : afrag_gf(ar, c - 4, lane);
#pragma unroll
      for (int t = 0; t < 8; ++t)
        acc[t] = wmma16(af, *bfrag_ptr(w1, c, t, lane), acc[t]);
    }
#pragma unroll
    for (int t = 0; t < 8; ++t)
#pragma unroll
      for (int r = 0; r < 8; ++r)
        tile[(r + hi8) * HF + t * 16 + m] = (h16)silu_f(acc[t][r]);
    asm volatile("s_wait_dscnt 0x0" ::: "memory");

#pragma unroll
    for (int t = 0; t < 8; ++t)
#pragma unroll
      for (int r = 0; r < 8; ++r) acc[t][r] = 0.0f;
#pragma unroll
    for (int c = 0; c < 4; ++c) {
      const v16h am = afrag_lds(tile, c, lane);
#pragma unroll
      for (int t = 0; t < 8; ++t)
        acc[t] = wmma16(am, *bfrag_ptr(w2, c, t, lane), acc[t]);
    }
#pragma unroll
    for (int t = 0; t < 8; ++t)
#pragma unroll
      for (int r = 0; r < 8; ++r)
        tileF[(r + hi8) * HF + t * 16 + m] = acc[t][r];
    asm volatile("s_wait_dscnt 0x0" ::: "memory");

    for (int idx = lane; idx < 16 * HF; idx += 32) {
      const int row = idx >> 7;
      const int nn  = ti * 16 + row;
      if (nn < nN) {
        const int k = idx & 127;
        const float v = hbuf[(size_t)nn * HF + k] + tileF[idx] + bh2[k];
        hbuf[(size_t)nn * HF + k]  = v;
        hhalf[(size_t)nn * HF + k] = (h16)v;
      }
    }
  }
}

// ---------------------------------------------------------------------------
// Host driver
// ---------------------------------------------------------------------------
extern "C" void kernel_launch(void* const* d_in, const int* in_sizes, int n_in,
                              void* d_out, int out_size, void* d_ws, size_t ws_size,
                              hipStream_t stream) {
  const int*   atom   = (const int*)d_in[0];
  const float* temb   = (const float*)d_in[1];
  const float* masses = (const float*)d_in[2];
  const float* mnorm  = (const float*)d_in[3];
  const float* clab   = (const float*)d_in[4];
  const float* cmask  = (const float*)d_in[5];
  const float* mom    = (const float*)d_in[6];
  const float* coords = (const float*)d_in[7];
  const int*   src    = (const int*)d_in[8];
  const int*   dst    = (const int*)d_in[9];
  const float* embed  = (const float*)d_in[10];
  const float* Wh     = (const float*)d_in[11];
  const float* bh     = (const float*)d_in[12];
  const float* We1    = (const float*)d_in[13];
  const float* be1    = (const float*)d_in[14];
  const float* We2    = (const float*)d_in[15];
  const float* be2    = (const float*)d_in[16];
  const float* Wx1    = (const float*)d_in[17];
  const float* bx1    = (const float*)d_in[18];
  const float* Wx2    = (const float*)d_in[19];
  const float* Wh1    = (const float*)d_in[20];
  const float* bh1    = (const float*)d_in[21];
  const float* Wh2    = (const float*)d_in[22];
  const float* bh2    = (const float*)d_in[23];
  (void)n_in; (void)out_size; (void)ws_size;

  const int N = in_sizes[0];
  const int E = in_sizes[8];

  // carve workspace
  char* ws = (char*)d_ws;
  size_t cur = 0;
  auto carve = [&](size_t bytes) -> char* {
    size_t o = (cur + 255) & ~(size_t)255;
    cur = o + bytes;
    return ws + o;
  };
  float* hbuf  = (float*)carve((size_t)N * HF * 4);
  h16*   hhalf = (h16*)  carve((size_t)N * HF * 2);
  float* agg   = (float*)carve((size_t)N * HF * 4);
  float* aBuf  = (float*)carve((size_t)E * 4);
  float* x0    = (float*)carve((size_t)N * 3 * 4);
  float* x1    = (float*)carve((size_t)N * 3 * 4);
  h16*   wfE   = (h16*)  carve((size_t)2 * 4 * 16384 * 2);
  h16*   wfN1  = (h16*)  carve((size_t)32768 * 2);
  h16*   wfN2  = (h16*)  carve((size_t)16384 * 2);

  hipFuncSetAttribute((const void*)edge_msg_kernel,
                      hipFuncAttributeMaxDynamicSharedMemorySize, EDGE_SMEM);
  hipFuncSetAttribute((const void*)node_update_kernel,
                      hipFuncAttributeMaxDynamicSharedMemorySize, NODE_SMEM);

  // 1) weight fragments
  for (int l = 0; l < 2; ++l) {
    frag_convert_kernel<<<64, 256, 0, stream>>>(We1 + (size_t)l * 257 * HF,
                                                wfE + ((size_t)l * 4 + 0) * 16384, 128);
    frag_convert_kernel<<<64, 256, 0, stream>>>(We1 + (size_t)l * 257 * HF + 128 * HF,
                                                wfE + ((size_t)l * 4 + 1) * 16384, 128);
    frag_convert_kernel<<<64, 256, 0, stream>>>(We2 + (size_t)l * 128 * HF,
                                                wfE + ((size_t)l * 4 + 2) * 16384, 128);
    frag_convert_kernel<<<64, 256, 0, stream>>>(Wx1 + (size_t)l * 128 * HF,
                                                wfE + ((size_t)l * 4 + 3) * 16384, 128);
  }
  frag_convert_kernel<<<128, 256, 0, stream>>>(Wh1, wfN1, 256);  // layer 0 only
  frag_convert_kernel<<<64, 256, 0, stream>>>(Wh2, wfN2, 128);

  // 2) node embedding projection, 3) edge distances, x/agg init
  node_init_kernel<<<N, 128, 0, stream>>>(atom, temb, masses, mnorm, clab, cmask,
                                          mom, embed, Wh, bh, hbuf, hhalf, N);
  edge_dist_kernel<<<(E + 255) / 256, 256, 0, stream>>>(coords, src, dst, aBuf, E);
  hipMemcpyAsync(x0, coords, (size_t)N * 3 * 4, hipMemcpyDeviceToDevice, stream);
  hipMemsetAsync(agg, 0, (size_t)N * HF * 4, stream);

  // 4) layers
  float* xc = x0;
  float* xn = x1;
  for (int l = 0; l < 2; ++l) {
    hipMemcpyAsync(xn, xc, (size_t)N * 3 * 4, hipMemcpyDeviceToDevice, stream);
    edge_msg_kernel<<<1024, 128, EDGE_SMEM, stream>>>(
        hhalf, xc, xn, agg, src, dst, aBuf, wfE + (size_t)l * 4 * 16384,
        be1 + (size_t)l * HF, be2 + (size_t)l * HF, bx1 + (size_t)l * HF,
        We1 + ((size_t)l * 257 + 256) * HF, Wx2 + (size_t)l * HF, E,
        (l == 0) ? 1 : 0);
    { float* t = xc; xc = xn; xn = t; }
    if (l == 0) {
      node_update_kernel<<<512, 128, NODE_SMEM, stream>>>(hbuf, hhalf, agg, wfN1,
                                                          wfN2, bh1, bh2, N);
    }
  }

  // 5) output
  hipMemcpyAsync(d_out, xc, (size_t)N * 3 * 4, hipMemcpyDeviceToDevice, stream);
}
